// InvKin_58102317580984
// MI455X (gfx1250) — matrix operand order repaired and checked by reading, not compiled, not run on hardware
//
#include <hip/hip_runtime.h>

// ---------------------------------------------------------------------------
// Types
// ---------------------------------------------------------------------------
typedef __bf16 bf16_t;
typedef __attribute__((ext_vector_type(16))) __bf16 v16bf;
typedef __attribute__((ext_vector_type(8)))  __bf16 v8bf;
typedef __attribute__((ext_vector_type(4)))  __bf16 v4bf;
typedef __attribute__((ext_vector_type(8)))  float  v8f;

#define HDIM 1024
#define TM 128
#define TN 128
#define TK 64
#define KTILES (HDIM / TK)   // 16
#define LDSS 88   // bf16 elements per LDS row (176B: 16B aligned, conflict-free)

// Use the Tensor Data Mover when the toolchain exposes it (device pass only).
#if defined(__HIP_DEVICE_COMPILE__) && __has_builtin(__builtin_amdgcn_tensor_load_to_lds)
#define IK_USE_TDM 1
#else
#define IK_USE_TDM 0
#endif

// ---------------------------------------------------------------------------
// Build a 16x32 bf16 A-fragment / 32x16 B-fragment from two 16B LDS chunks
// ---------------------------------------------------------------------------
static __device__ inline v16bf make_frag(const bf16_t* p_lo, const bf16_t* p_hi) {
    v8bf lo = *(const v8bf*)p_lo;
    v8bf hi = *(const v8bf*)p_hi;
    v16bf r;
#pragma unroll
    for (int i = 0; i < 8; ++i) { r[i] = lo[i]; r[i + 8] = hi[i]; }
    return r;
}

// One K-slab (TK wide) of WMMA work for this wave: 4x2 fragments, 2 k-steps.
static __device__ inline void ik_compute_slab(const bf16_t* Asb, const bf16_t* Bsb,
                                              v8f acc[4][2], int wm0, int wn0,
                                              int g, int ln) {
#pragma unroll
    for (int ks = 0; ks < TK; ks += 32) {
        v16bf afrag[4];
#pragma unroll
        for (int mf = 0; mf < 4; ++mf) {
            const bf16_t* p = Asb + (wm0 + mf * 16 + ln) * LDSS + ks + g * 8;
            afrag[mf] = make_frag(p, p + 16);   // K = g*8+[0,8) and 16+g*8+[0,8)
        }
        v16bf bfrag[2];
#pragma unroll
        for (int nf = 0; nf < 2; ++nf) {
            const bf16_t* p = Bsb + (wn0 + nf * 16 + ln) * LDSS + ks + g * 8;
            bfrag[nf] = make_frag(p, p + 16);   // K = g*8+[0,8) and 16+g*8+[0,8)
        }
#pragma unroll
        for (int mf = 0; mf < 4; ++mf)
#pragma unroll
            for (int nf = 0; nf < 2; ++nf)
                acc[mf][nf] = __builtin_amdgcn_wmma_f32_16x16x32_bf16(
                    false, afrag[mf], false, bfrag[nf],
                    (short)0, acc[mf][nf], false, false);
    }
}

#if IK_USE_TDM
typedef __attribute__((address_space(3))) const char* ik_lds_cptr;
static __device__ inline unsigned ik_lds_addr(const void* p) {
    return (unsigned)(unsigned long long)(ik_lds_cptr)p;
}

// TDM descriptor per CDNA5 ISA:
// 2D tensor, bf16 (data_size=2B), tile TK x TM, row stride HDIM elements,
// LDS padding: 32 DWORDs (one 64-elem row) -> +12 DWORDs (row stride 88).
static __device__ inline void ik_tdm_load_tile(unsigned lds_off, const void* gptr,
                                               unsigned nrows) {
    typedef __attribute__((ext_vector_type(4))) unsigned int u32x4;
    typedef __attribute__((ext_vector_type(8))) int i32x8;
    typedef __attribute__((ext_vector_type(4))) int i32x4;
    unsigned long long ga = (unsigned long long)(unsigned long long)(size_t)gptr;
    u32x4 g0;
    g0[0] = 1u;                                            // count=1, user mode
    g0[1] = lds_off;                                       // lds_addr (bytes)
    g0[2] = (unsigned)ga;                                  // global_addr[31:0]
    g0[3] = (unsigned)((ga >> 32) & 0x01FFFFFFu) | (2u << 30);  // [56:32] + type=2
    i32x8 g1;
    g1[0] = (1 << 16)      // data_size = 1 -> 2 bytes
          | (1 << 20)      // pad_enable
          | (4 << 22)      // pad_interval: code 4 = 32 DWORDs (one tile row)
          | (11 << 25);    // pad_amount:   code 11 = 12 DWORDs (24 bf16)
    g1[1] = (int)((HDIM & 0xFFFFu) << 16);                              // tensor_dim0 lo
    g1[2] = (int)(((HDIM >> 16) & 0xFFFFu) | ((nrows & 0xFFFFu) << 16)); // dim0 hi | dim1 lo
    g1[3] = (int)(((nrows >> 16) & 0xFFFFu) | ((unsigned)TK << 16));     // dim1 hi | tile_dim0
    g1[4] = TM;                                            // tile_dim1=128, tile_dim2=0
    g1[5] = HDIM;                                          // tensor_dim0_stride lo
    g1[6] = 0;
    g1[7] = 0;
    i32x4 gz = {0, 0, 0, 0};
#if __clang_major__ >= 23
    i32x8 gz8 = {0, 0, 0, 0, 0, 0, 0, 0};
    __builtin_amdgcn_tensor_load_to_lds(g0, g1, gz, gz, gz8, 0);
#else
    __builtin_amdgcn_tensor_load_to_lds(g0, g1, gz, gz, 0);
#endif
}
#endif  // IK_USE_TDM

// ---------------------------------------------------------------------------
// W (K-major f32) -> Wt (N-major bf16)
// ---------------------------------------------------------------------------
__global__ __launch_bounds__(256) void ik_convertT_kernel(
    const float* __restrict__ W, bf16_t* __restrict__ Wt) {
    int idx = blockIdx.x * 256 + threadIdx.x;          // over 1024*1024
    int n = idx >> 10;
    int k = idx & (HDIM - 1);
    Wt[idx] = (bf16_t)W[(size_t)k * HDIM + n];
}

// ---------------------------------------------------------------------------
// Layer 1: h = x(B,3) @ W1(3,1024) + b1 ; deterministic BN partials
// ---------------------------------------------------------------------------
__global__ __launch_bounds__(256) void ik_layer1_kernel(
    const float* __restrict__ x, const float* __restrict__ W1,
    const float* __restrict__ b1, float* __restrict__ Out,
    float2* __restrict__ partials) {
    __shared__ float sx[128][3];
    __shared__ float sW[3][HDIM];
    __shared__ float sb[HDIM];
    const int tid  = threadIdx.x;
    const int row0 = blockIdx.x * 128;
    for (int i = tid; i < 384;  i += 256) sx[i / 3][i % 3] = x[(size_t)row0 * 3 + i];
    for (int i = tid; i < 3072; i += 256) sW[i >> 10][i & (HDIM - 1)] = W1[i];
    for (int i = tid; i < HDIM; i += 256) sb[i] = b1[i];
    __syncthreads();

    const int c0 = tid * 4;
    float4 w0 = *(const float4*)&sW[0][c0];
    float4 w1 = *(const float4*)&sW[1][c0];
    float4 w2 = *(const float4*)&sW[2][c0];
    float4 bb = *(const float4*)&sb[c0];
    float4 s = {0, 0, 0, 0}, q = {0, 0, 0, 0};
    for (int r = 0; r < 128; ++r) {
        float x0 = sx[r][0], x1 = sx[r][1], x2 = sx[r][2];
        float4 h;
        h.x = fmaf(x0, w0.x, fmaf(x1, w1.x, fmaf(x2, w2.x, bb.x)));
        h.y = fmaf(x0, w0.y, fmaf(x1, w1.y, fmaf(x2, w2.y, bb.y)));
        h.z = fmaf(x0, w0.z, fmaf(x1, w1.z, fmaf(x2, w2.z, bb.z)));
        h.w = fmaf(x0, w0.w, fmaf(x1, w1.w, fmaf(x2, w2.w, bb.w)));
        *(float4*)&Out[(size_t)(row0 + r) * HDIM + c0] = h;
        s.x += h.x; q.x += h.x * h.x;
        s.y += h.y; q.y += h.y * h.y;
        s.z += h.z; q.z += h.z * h.z;
        s.w += h.w; q.w += h.w * h.w;
    }
    float2* p = partials + (size_t)blockIdx.x * HDIM + c0;
    p[0] = make_float2(s.x, q.x);
    p[1] = make_float2(s.y, q.y);
    p[2] = make_float2(s.z, q.z);
    p[3] = make_float2(s.w, q.w);
}

// ---------------------------------------------------------------------------
// Finalize BN stats (fixed-order -> deterministic):
// stats[col] = (scale, shift) so act = relu(h*scale + shift)
// ---------------------------------------------------------------------------
__global__ __launch_bounds__(256) void ik_stats_kernel(
    const float2* __restrict__ partials, const float* __restrict__ gg,
    const float* __restrict__ bt, float2* __restrict__ stats,
    float invB, int nchunks) {
    int col = blockIdx.x * 256 + threadIdx.x;
    if (col >= HDIM) return;
    float S = 0.f, Q = 0.f;
    for (int t = 0; t < nchunks; ++t) {
        float2 p = partials[(size_t)t * HDIM + col];
        S += p.x; Q += p.y;
    }
    float mu   = S * invB;
    float var  = fmaxf(Q * invB - mu * mu, 0.f);
    float rstd = rsqrtf(var + 1e-5f);
    float sc   = gg[col] * rstd;
    stats[col] = make_float2(sc, bt[col] - mu * sc);
}

// ---------------------------------------------------------------------------
// BN + ReLU + cast to bf16 (streaming, float4 in / 4x bf16 out)
// ---------------------------------------------------------------------------
__global__ __launch_bounds__(256) void ik_act_kernel(
    const float* __restrict__ Hpre, const float2* __restrict__ stats,
    bf16_t* __restrict__ Act, long long n4) {
    long long i = (long long)blockIdx.x * 256 + threadIdx.x;
    if (i >= n4) return;
    size_t e  = (size_t)i * 4;
    int   cb  = (int)(e & (HDIM - 1));
    float4 h  = *(const float4*)(Hpre + e);
    float2 s0 = stats[cb], s1 = stats[cb + 1], s2 = stats[cb + 2], s3 = stats[cb + 3];
    v4bf o;
    o[0] = (bf16_t)fmaxf(fmaf(h.x, s0.x, s0.y), 0.f);
    o[1] = (bf16_t)fmaxf(fmaf(h.y, s1.x, s1.y), 0.f);
    o[2] = (bf16_t)fmaxf(fmaf(h.z, s2.x, s2.y), 0.f);
    o[3] = (bf16_t)fmaxf(fmaf(h.w, s3.x, s3.y), 0.f);
    *(v4bf*)(Act + e) = o;
}

// ---------------------------------------------------------------------------
// WMMA GEMM: Out(B,1024) = A(B,1024 bf16) @ Wt^T + bias, fused BN partials.
// Block = 128x128 tile, 8 waves, 64x32 per wave (4x2 wmma accumulators).
// TDM path: double-buffered LDS, wave 0 drives tensor_load_to_lds DMA for the
// next K-slab while all waves run WMMA on the current one (TENSORcnt overlap).
// ---------------------------------------------------------------------------
__global__ __launch_bounds__(256) void ik_gemm_bn_kernel(
    const bf16_t* __restrict__ A,   // [arows][1024] row-major
    const bf16_t* __restrict__ Wt,  // [1024 n][1024 k] N-major
    const float*  __restrict__ bias,
    float*  __restrict__ Out,       // [arows][1024]
    float2* __restrict__ partials,  // [arows/128][1024]
    int arows)
{
    __shared__ bf16_t As[2][TM * LDSS];
    __shared__ bf16_t Bs[2][TN * LDSS];
    __shared__ float  s_sum[4][TN];
    __shared__ float  s_sq[4][TN];

    const int tid    = threadIdx.x;
    const int wid    = tid >> 5;
    const int lane   = tid & 31;
    const int g      = lane >> 4;   // lane group (K-half selector)
    const int ln     = lane & 15;   // row (A) / col (B) within fragment
    const int wave_m = wid & 1;
    const int wave_n = wid >> 1;
    const int wm0    = wave_m * 64;
    const int wn0    = wave_n * 32;
    const int row0   = blockIdx.x * TM;
    const int col0   = blockIdx.y * TN;

    v8f zero = {};
    v8f acc[4][2];
#pragma unroll
    for (int mf = 0; mf < 4; ++mf)
#pragma unroll
        for (int nf = 0; nf < 2; ++nf) acc[mf][nf] = zero;

#if IK_USE_TDM
    const bf16_t* Abase = A  + (size_t)row0 * HDIM;
    const bf16_t* Bbase = Wt + (size_t)col0 * HDIM;
    if (wid == 0) {   // prologue: DMA slab 0 into buffer 0
        ik_tdm_load_tile(ik_lds_addr(&As[0][0]), Abase, (unsigned)arows);
        ik_tdm_load_tile(ik_lds_addr(&Bs[0][0]), Bbase, HDIM);
    }
    for (int kt = 0; kt < KTILES; ++kt) {
        const int cur = kt & 1;
        if ((kt + 1 < KTILES) && wid == 0) {   // DMA next slab into other buffer
            ik_tdm_load_tile(ik_lds_addr(&As[cur ^ 1][0]),
                             Abase + (kt + 1) * TK, (unsigned)arows);
            ik_tdm_load_tile(ik_lds_addr(&Bs[cur ^ 1][0]),
                             Bbase + (kt + 1) * TK, HDIM);
        }
        if (kt + 1 < KTILES) __builtin_amdgcn_s_wait_tensorcnt(2);  // cur done, next in flight
        else                 __builtin_amdgcn_s_wait_tensorcnt(0);
        __syncthreads();                        // cur slab visible to all waves
        ik_compute_slab(&As[cur][0], &Bs[cur][0], acc, wm0, wn0, g, ln);
        __syncthreads();                        // all reads done before reuse by DMA
    }
#else
    for (int kt = 0; kt < KTILES; ++kt) {
        const bf16_t* Ag = A  + (size_t)row0 * HDIM + kt * TK;
        const bf16_t* Bg = Wt + (size_t)col0 * HDIM + kt * TK;
        if (kt + 1 < KTILES) {  // prefetch next K slab -> global_prefetch_b8
            __builtin_prefetch(Ag + (size_t)(tid >> 1) * HDIM + TK + (tid & 1) * 32, 0, 1);
            __builtin_prefetch(Bg + (size_t)(tid >> 1) * HDIM + TK + (tid & 1) * 32, 0, 1);
        }
        __syncthreads();
#pragma unroll
        for (int i = 0; i < 4; ++i) {           // 1024 16B chunks / 256 threads
            int c = tid + i * 256;
            int r = c >> 3, c8 = c & 7;
            *(v8bf*)(&As[0][0] + r * LDSS + c8 * 8) =
                *(const v8bf*)(Ag + (size_t)r * HDIM + c8 * 8);
            *(v8bf*)(&Bs[0][0] + r * LDSS + c8 * 8) =
                *(const v8bf*)(Bg + (size_t)r * HDIM + c8 * 8);
        }
        __syncthreads();
        ik_compute_slab(&As[0][0], &Bs[0][0], acc, wm0, wn0, g, ln);
    }
#endif

    // Epilogue: bias add, store f32 pre-activation, deterministic BN partials
    float s_loc[2] = {0.f, 0.f}, q_loc[2] = {0.f, 0.f};
#pragma unroll
    for (int nf = 0; nf < 2; ++nf) {
        int col  = col0 + wn0 + nf * 16 + ln;
        float bv = bias[col];
#pragma unroll
        for (int mf = 0; mf < 4; ++mf) {
            int rbase = row0 + wm0 + mf * 16 + g * 8;
#pragma unroll
            for (int v = 0; v < 8; ++v) {
                float val = acc[mf][nf][v] + bv;
                Out[(size_t)(rbase + v) * HDIM + col] = val;
                s_loc[nf] += val;
                q_loc[nf] += val * val;
            }
        }
    }
    const int slot = wave_m * 2 + g;   // 4 unique contributors per tile column
#pragma unroll
    for (int nf = 0; nf < 2; ++nf) {
        int cl = wn0 + nf * 16 + ln;
        s_sum[slot][cl] = s_loc[nf];
        s_sq[slot][cl]  = q_loc[nf];
    }
    __syncthreads();
    if (tid < TN) {
        float S = s_sum[0][tid] + s_sum[1][tid] + s_sum[2][tid] + s_sum[3][tid];
        float Q = s_sq[0][tid]  + s_sq[1][tid]  + s_sq[2][tid]  + s_sq[3][tid];
        partials[(size_t)blockIdx.x * HDIM + col0 + tid] = make_float2(S, Q);
    }
}

// ---------------------------------------------------------------------------
// Final: BN3+ReLU on the fly, thetas = h3 @ W4 + b4, forward kinematics.
// One wave (32 lanes) per batch row; 8 rows per block.
// ---------------------------------------------------------------------------
__global__ __launch_bounds__(256) void ik_final_kernel(
    const float* __restrict__ Hpre, const float2* __restrict__ stats,
    const float* __restrict__ W4, const float* __restrict__ b4,
    float* __restrict__ out, int Brows) {
    __shared__ float  sW[HDIM * 3];
    __shared__ float2 sStat[HDIM];
    const int tid = threadIdx.x;
    for (int i = tid; i < HDIM * 3; i += 256) sW[i] = W4[i];
    for (int i = tid; i < HDIM;     i += 256) sStat[i] = stats[i];
    __syncthreads();

    const int wid = tid >> 5, lane = tid & 31;
    const int r   = blockIdx.x * 8 + wid;
    const float* hrow = Hpre + (size_t)r * HDIM;
    float t0 = 0.f, t1 = 0.f, t2 = 0.f;
#pragma unroll
    for (int ch = 0; ch < 8; ++ch) {
        int k = ch * 128 + lane * 4;
        float4 h = *(const float4*)(hrow + k);
        float hv[4] = {h.x, h.y, h.z, h.w};
#pragma unroll
        for (int j = 0; j < 4; ++j) {
            float2 st = sStat[k + j];
            float  a  = fmaxf(fmaf(hv[j], st.x, st.y), 0.f);
            const float* w = sW + (size_t)(k + j) * 3;
            t0 = fmaf(a, w[0], t0);
            t1 = fmaf(a, w[1], t1);
            t2 = fmaf(a, w[2], t2);
        }
    }
#pragma unroll
    for (int off = 16; off > 0; off >>= 1) {   // wave32 reduction
        t0 += __shfl_down(t0, off);
        t1 += __shfl_down(t1, off);
        t2 += __shfl_down(t2, off);
    }
    if (lane == 0) {
        float th0 = t0 + b4[0], th1 = t1 + b4[1], th2 = t2 + b4[2];
        out[(size_t)r * 3 + 0] = th0;
        out[(size_t)r * 3 + 1] = th1;
        out[(size_t)r * 3 + 2] = th2;
        const float R1 = 0.12f, R2 = 0.115f;
        float c1 = cosf(th1), s1 = sinf(th1);
        float c12 = cosf(th1 + th2), s12 = sinf(th1 + th2);
        float L = fmaf(R2, c12, R1 * c1);
        size_t base = (size_t)Brows * 3 + (size_t)r * 3;
        out[base + 0] = cosf(th0) * L;
        out[base + 1] = sinf(th0) * L;
        out[base + 2] = fmaf(R2, s12, R1 * s1);
    }
}

// ---------------------------------------------------------------------------
// Host launcher
// ---------------------------------------------------------------------------
extern "C" void kernel_launch(void* const* d_in, const int* in_sizes, int n_in,
                              void* d_out, int out_size, void* d_ws, size_t ws_size,
                              hipStream_t stream) {
    const int B = in_sizes[0] / 3;   // 131072
    const float* x   = (const float*)d_in[0];
    const float* W1  = (const float*)d_in[1];
    const float* b1  = (const float*)d_in[2];
    const float* g1  = (const float*)d_in[3];
    const float* bt1 = (const float*)d_in[4];
    const float* W2  = (const float*)d_in[5];
    const float* b2  = (const float*)d_in[6];
    const float* g2  = (const float*)d_in[7];
    const float* bt2 = (const float*)d_in[8];
    const float* W3  = (const float*)d_in[9];
    const float* b3  = (const float*)d_in[10];
    const float* g3  = (const float*)d_in[11];
    const float* bt3 = (const float*)d_in[12];
    const float* W4  = (const float*)d_in[13];
    const float* b4  = (const float*)d_in[14];

    char* ws = (char*)d_ws;
    size_t off = 0;
    float*  h_pre = (float*)(ws + off);  off += (size_t)B * HDIM * sizeof(float);
    bf16_t* h_act = (bf16_t*)(ws + off); off += (size_t)B * HDIM * sizeof(bf16_t);
    bf16_t* wt2   = (bf16_t*)(ws + off); off += (size_t)HDIM * HDIM * sizeof(bf16_t);
    bf16_t* wt3   = (bf16_t*)(ws + off); off += (size_t)HDIM * HDIM * sizeof(bf16_t);
    float2* parts = (float2*)(ws + off); off += (size_t)(B / 128) * HDIM * sizeof(float2);
    float2* stats = (float2*)(ws + off); off += (size_t)HDIM * sizeof(float2);

    const float invB = 1.0f / (float)B;
    const int nchunks = B / 128;                 // 1024
    const long long n4 = (long long)B * HDIM / 4;
    const dim3 gemm_grid(B / TM, HDIM / TN);     // (1024, 8)

    ik_convertT_kernel<<<HDIM * HDIM / 256, 256, 0, stream>>>(W2, wt2);
    ik_convertT_kernel<<<HDIM * HDIM / 256, 256, 0, stream>>>(W3, wt3);

    // Layer 1 (K=3, VALU) + BN1
    ik_layer1_kernel<<<B / 128, 256, 0, stream>>>(x, W1, b1, h_pre, parts);
    ik_stats_kernel<<<4, 256, 0, stream>>>(parts, g1, bt1, stats, invB, nchunks);
    ik_act_kernel<<<(unsigned)(n4 / 256), 256, 0, stream>>>(h_pre, stats, h_act, n4);

    // Layer 2 (WMMA) + BN2
    ik_gemm_bn_kernel<<<gemm_grid, 256, 0, stream>>>(h_act, wt2, b2, h_pre, parts, B);
    ik_stats_kernel<<<4, 256, 0, stream>>>(parts, g2, bt2, stats, invB, nchunks);
    ik_act_kernel<<<(unsigned)(n4 / 256), 256, 0, stream>>>(h_pre, stats, h_act, n4);

    // Layer 3 (WMMA) + BN3
    ik_gemm_bn_kernel<<<gemm_grid, 256, 0, stream>>>(h_act, wt3, b3, h_pre, parts, B);
    ik_stats_kernel<<<4, 256, 0, stream>>>(parts, g3, bt3, stats, invB, nchunks);

    // Layer 4 + kinematics, fused with BN3 normalization
    ik_final_kernel<<<B / 8, 256, 0, stream>>>(h_pre, stats, W4, b4, (float*)d_out, B);
}
